// MultiViewGCN_45097156608543
// MI455X (gfx1250) — matrix operand order repaired and checked by reading, not compiled, not run on hardware
//
#include <hip/hip_runtime.h>
#include <hip/hip_bf16.h>

// ---------------------------------------------------------------------------
// MultiViewGCN for MI455X (gfx1250, wave32, WMMA).
//
// Roofline: dominant op is Anorm @ [Xf|Xs] = 8192x8192x512 -> 68.7 GFLOP.
// adj read once = 256 MB = ~11us @ 23.3 TB/s; bf16 WMMA compute ~15-30us
// (from 20.1 PTOPS fp8 sparse -> ~5 PFLOPS bf16 dense) -> near the
// compute/memory crossover, so A must be read from HBM only ONCE:
//   * grid is launched N-fastest so the 4 blocks that share a 128-row A band
//     (4 MB, L2-resident in 192 MB) are scheduled concurrently.
//   * A lines are prefetched 4 K-steps ahead, UNCONDITIONALLY: SE-scope
//     global_prefetch_b8 is speculative (translation failures silently
//     dropped), so no bounds branch is needed in the hot loop.
// v_wmma_f32_16x16x32_bf16 gives 8x the FLOP/instr of the f32 16x16x4 op;
// bf16 quantization of U(0,1) adjacency with f32 accumulation keeps ~1e-2
// relative error on msg (which only feeds a sigmoid gate + blend).
//
// Algebra: diag(dis) A diag(dis) @ X == diag(dis) (A @ (diag(dis) X)), so the
// 256MB adjacency is consumed raw (f32 -> bf16 in registers), never rescaled.
//
// No LDS/TDM staging: the 4 b128 loads per A fragment consume whole 128B
// lines (100% utilization), B (8 MB) is L2-resident, and async double
// buffering would cost 2 workgroup barriers per K-step (256 iters).
//
// All 4 B fragments are loaded into distinct registers BEFORE the 8 WMMAs of
// a K-step so the scheduler can clause the loads and retire them with partial
// s_wait_loadcnt while earlier WMMAs execute.
//
// WMMA fragment layout assumptions (from CDNA5 ISA 7.12.2, wave32):
//   A (16x32 bf16): lane&15 = M row; laneHalf s = (lane>=16)*8;
//                   vector elems 0..7  = K s..s+7, elems 8..15 = K s+16..s+23.
//   B (32x16 bf16): lane&15 = N col; elems 0..7 = K s..s+7, elems 8..15 = K s+16..s+23
//                   (per the B-matrix text in ISA 7.12.4).
//   C/D (16x16 f32): elem r -> M = r + 8*laneHalf, N = lane&15.
// ---------------------------------------------------------------------------

typedef __bf16 v16bf __attribute__((ext_vector_type(16)));
typedef __bf16 v8bf  __attribute__((ext_vector_type(8)));
typedef float  v8f   __attribute__((ext_vector_type(8)));
typedef float  v4f   __attribute__((ext_vector_type(4)));

#define NN   8192   // nodes
#define IND  256    // input dim
#define HID  256    // hidden dim
#define XSC  512    // concatenated X columns (2 views)

// --------------------------- kernel 1: row sums -----------------------------
__global__ void k_rowsum(const float* __restrict__ adj, float* __restrict__ dis)
{
    const int row = blockIdx.x;
    const int tid = threadIdx.x;
    const v4f* r4 = (const v4f*)(adj + (size_t)row * NN);
    float s = 0.f;
    #pragma unroll 4
    for (int c = tid; c < NN / 4; c += 256) {
        v4f v = r4[c];
        s += v[0] + v[1] + v[2] + v[3];
    }
    __shared__ float red[256];
    red[tid] = s;
    __syncthreads();
    for (int off = 128; off > 0; off >>= 1) {
        if (tid < off) red[tid] += red[tid + off];
        __syncthreads();
    }
    if (tid == 0) dis[row] = rsqrtf(red[0] + 1e-9f);
}

// ------------------ kernel 2: build bf16 Xs (=dis*X) and bf16 W -------------
__global__ void k_scale_inputs(const float* __restrict__ fx, const float* __restrict__ sx,
                               const float* __restrict__ Wf, const float* __restrict__ Ws,
                               const float* __restrict__ dis,
                               __bf16* __restrict__ Xs,
                               __bf16* __restrict__ Wfb, __bf16* __restrict__ Wsb)
{
    const int j = blockIdx.x;
    const int c = threadIdx.x;
    const float d = dis[j];
    Xs[(size_t)j * XSC + c]       = (__bf16)(d * fx[(size_t)j * IND + c]);
    Xs[(size_t)j * XSC + 256 + c] = (__bf16)(d * sx[(size_t)j * IND + c]);
    if (j < IND) {
        Wfb[j * HID + c] = (__bf16)Wf[j * HID + c];
        Wsb[j * HID + c] = (__bf16)Ws[j * HID + c];
    }
}

// ---------------- kernel 3: msg = diag(dis) * (A @ Xs)  (bf16 WMMA) ---------
// Grid (4, 64): x = N group (fastest -> concurrent blocks share an A band),
//               y = M band. Block 256 = 8 waves, block tile 128(M) x 128(N).
// Wave tile 32(M) x 64(N): 2x4 accumulators of 16x16.
__global__ void __launch_bounds__(256)
k_gemm_adj(const float* __restrict__ A, const __bf16* __restrict__ Xs,
           const float* __restrict__ dis, __bf16* __restrict__ msg)
{
    const int lane = threadIdx.x & 31;
    const int wid  = threadIdx.x >> 5;
    const int half = lane >> 4;       // lane-half: selects K sub-window
    const int l16  = lane & 15;
    const int m_wave = blockIdx.y * 128 + (wid >> 1) * 32;
    const int n_wave = blockIdx.x * 128 + (wid & 1) * 64;

    v8f acc[2][4] = {};

    for (int k0 = 0; k0 < NN; k0 += 32) {
        // ---- A fragments (f32 -> bf16 in registers) ----
        v16bf afrag[2];
        #pragma unroll
        for (int im = 0; im < 2; ++im) {
            const float* ar = A + (size_t)(m_wave + im * 16 + l16) * NN + k0 + half * 8;
            v4f a0 = *(const v4f*)(ar + 0);    // K = s .. s+3
            v4f a1 = *(const v4f*)(ar + 4);    // K = s+4 .. s+7
            v4f a2 = *(const v4f*)(ar + 16);   // K = s+16 .. s+19
            v4f a3 = *(const v4f*)(ar + 20);   // K = s+20 .. s+23
            v16bf f;
            #pragma unroll
            for (int t = 0; t < 4; ++t) {
                f[t]      = (__bf16)a0[t];
                f[4 + t]  = (__bf16)a1[t];
                f[8 + t]  = (__bf16)a2[t];
                f[12 + t] = (__bf16)a3[t];
            }
            afrag[im] = f;
            // Speculative prefetch 4 K-steps ahead (128B line = one 32-K
            // window). Unconditional: OOB prefetches are dropped by HW.
            __builtin_prefetch(ar + 128, 0, 3);
        }
        // ---- B fragments: load all four into distinct registers first ----
        v16bf bfrag[4];
        #pragma unroll
        for (int in = 0; in < 4; ++in) {
            const __bf16* br = Xs + (size_t)(k0 + half * 8) * XSC + n_wave + in * 16 + l16;
            v16bf bf;
            #pragma unroll
            for (int t = 0; t < 8; ++t) { bf[t] = br[(size_t)t * XSC]; bf[8 + t] = br[(size_t)(16 + t) * XSC]; }
            bfrag[in] = bf;
        }
        // ---- 8 WMMAs for this K-step ----
        #pragma unroll
        for (int in = 0; in < 4; ++in) {
            #pragma unroll
            for (int im = 0; im < 2; ++im) {
                acc[im][in] = __builtin_amdgcn_wmma_f32_16x16x32_bf16(
                    false, afrag[im], false, bfrag[in], (short)0, acc[im][in], false, false);
            }
        }
    }

    // ---- epilogue: row-scale by dis[m], store bf16 msg ----
    #pragma unroll
    for (int im = 0; im < 2; ++im) {
        #pragma unroll
        for (int r = 0; r < 8; ++r) {
            const int m = m_wave + im * 16 + r + half * 8;
            const float scale = dis[m];
            #pragma unroll
            for (int in = 0; in < 4; ++in) {
                const int n = n_wave + in * 16 + l16;
                msg[(size_t)m * XSC + n] = (__bf16)(acc[im][in][r] * scale);
            }
        }
    }
}

// -------- kernel 4: H_v = relu(msg_v @ W_v + b_v)  (bf16 WMMA, K=256) -------
// Grid (2, 64, 2): x = N group, y = M band, z = view.
__global__ void __launch_bounds__(256)
k_gemm_h(const __bf16* __restrict__ msg,
         const __bf16* __restrict__ Wfb, const __bf16* __restrict__ Wsb,
         const float* __restrict__ bf_, const float* __restrict__ bs_,
         float* __restrict__ Hf, float* __restrict__ Hs)
{
    const int view = blockIdx.z;
    const __bf16* W   = view ? Wsb : Wfb;
    const float* bias = view ? bs_ : bf_;
    float*       H    = view ? Hs  : Hf;

    const int lane = threadIdx.x & 31;
    const int wid  = threadIdx.x >> 5;
    const int half = lane >> 4;
    const int l16  = lane & 15;
    const int m_wave = blockIdx.y * 128 + (wid >> 1) * 32;
    const int n_wave = blockIdx.x * 128 + (wid & 1) * 64;

    v8f acc[2][4] = {};

    #pragma unroll
    for (int k0 = 0; k0 < IND; k0 += 32) {
        v16bf afrag[2];
        #pragma unroll
        for (int im = 0; im < 2; ++im) {
            const __bf16* ar = msg + (size_t)(m_wave + im * 16 + l16) * XSC
                             + view * 256 + k0 + half * 8;
            v8bf a0 = *(const v8bf*)ar;          // K = s .. s+7
            v8bf a1 = *(const v8bf*)(ar + 16);   // K = s+16 .. s+23
            v16bf f;
            #pragma unroll
            for (int t = 0; t < 8; ++t) { f[t] = a0[t]; f[8 + t] = a1[t]; }
            afrag[im] = f;
        }
        v16bf bfrag[4];
        #pragma unroll
        for (int in = 0; in < 4; ++in) {
            const __bf16* br = W + (size_t)(k0 + half * 8) * HID + n_wave + in * 16 + l16;
            v16bf bf;
            #pragma unroll
            for (int t = 0; t < 8; ++t) { bf[t] = br[t * HID]; bf[8 + t] = br[(16 + t) * HID]; }
            bfrag[in] = bf;
        }
        #pragma unroll
        for (int in = 0; in < 4; ++in) {
            #pragma unroll
            for (int im = 0; im < 2; ++im) {
                acc[im][in] = __builtin_amdgcn_wmma_f32_16x16x32_bf16(
                    false, afrag[im], false, bfrag[in], (short)0, acc[im][in], false, false);
            }
        }
    }

    #pragma unroll
    for (int in = 0; in < 4; ++in) {
        const int n = n_wave + in * 16 + l16;
        const float bval = bias[n];
        #pragma unroll
        for (int im = 0; im < 2; ++im) {
            #pragma unroll
            for (int r = 0; r < 8; ++r) {
                const int m = m_wave + im * 16 + r + half * 8;
                float v = acc[im][in][r] + bval;
                H[(size_t)m * HID + n] = v > 0.f ? v : 0.f;
            }
        }
    }
}

// --------- kernel 5: attn = sigmoid([Hf|Hs] @ Wa + ba); blend ---------------
__global__ void k_attend(const float* __restrict__ Hf, const float* __restrict__ Hs,
                         const float* __restrict__ Wa, const float* __restrict__ ba,
                         float* __restrict__ out)
{
    const int i = blockIdx.x;
    const int c = threadIdx.x;
    const float hf = Hf[(size_t)i * HID + c];
    const float hs = Hs[(size_t)i * HID + c];
    float p = hf * Wa[c] + hs * Wa[HID + c];

    __shared__ float red[256];
    __shared__ float a_sh;
    red[c] = p;
    __syncthreads();
    for (int off = 128; off > 0; off >>= 1) {
        if (c < off) red[c] += red[c + off];
        __syncthreads();
    }
    if (c == 0) a_sh = 1.f / (1.f + __expf(-(red[0] + ba[0])));
    __syncthreads();
    const float a = a_sh;
    out[(size_t)i * HID + c] = a * hf + (1.f - a) * hs;
}

// ---------------------------------------------------------------------------
extern "C" void kernel_launch(void* const* d_in, const int* in_sizes, int n_in,
                              void* d_out, int out_size, void* d_ws, size_t ws_size,
                              hipStream_t stream)
{
    const float* fx  = (const float*)d_in[0];
    const float* sx  = (const float*)d_in[1];
    const float* adj = (const float*)d_in[2];
    const float* Wf  = (const float*)d_in[3];
    const float* bf_ = (const float*)d_in[4];
    const float* Ws  = (const float*)d_in[5];
    const float* bs_ = (const float*)d_in[6];
    const float* Wa  = (const float*)d_in[7];
    const float* ba  = (const float*)d_in[8];

    // Workspace layout (all offsets 256B-aligned), total ~33.9 MB:
    char* ws = (char*)d_ws;
    float*  dis = (float*) (ws);                                         // 32 KB
    __bf16* Xs  = (__bf16*)(ws + 32768);                                 // 8 MB
    __bf16* msg = (__bf16*)(ws + 32768 + 8388608);                       // 8 MB
    __bf16* Wfb = (__bf16*)(ws + 32768 + 2 * 8388608);                   // 128 KB
    __bf16* Wsb = (__bf16*)(ws + 32768 + 2 * 8388608 + 131072);          // 128 KB
    float*  Hf  = (float*) (ws + 32768 + 2 * 8388608 + 2 * 131072);      // 8 MB
    float*  Hs  = (float*) (ws + 32768 + 3 * 8388608 + 2 * 131072);      // 8 MB

    k_rowsum      <<<NN, 256, 0, stream>>>(adj, dis);
    k_scale_inputs<<<NN, 256, 0, stream>>>(fx, sx, Wf, Ws, dis, Xs, Wfb, Wsb);
    k_gemm_adj    <<<dim3(4, 64),    256, 0, stream>>>(adj, Xs, dis, msg);
    k_gemm_h      <<<dim3(2, 64, 2), 256, 0, stream>>>(msg, Wfb, Wsb, bf_, bs_, Hf, Hs);
    k_attend      <<<NN, 256, 0, stream>>>(Hf, Hs, Wa, ba, (float*)d_out);
}
